// GraphModel_60756607369297
// MI455X (gfx1250) — matrix operand order repaired and checked by reading, not compiled, not run on hardware
//
#include <hip/hip_runtime.h>
#include <hip/hip_bf16.h>

typedef __bf16 bf16_t;
typedef __attribute__((ext_vector_type(16))) __bf16 v16bf;
typedef __attribute__((ext_vector_type(8)))  __bf16 v8bf;
typedef __attribute__((ext_vector_type(8)))  float  v8f;

// async-to-LDS builtin takes int4* in global (as1) and LDS (as3) address spaces
typedef __attribute__((ext_vector_type(4))) int v4i;
typedef __attribute__((address_space(1))) v4i gas_v4i;
typedef __attribute__((address_space(3))) v4i las_v4i;

#define F_IN 512
#define HID  512
#define DOUT 128
#define BN_EPS 1e-5f
#define BKC 128          // K-chunk staged in LDS per iteration
#define BN_ROWS 64       // rows per block in BN partial-sum kernel

__device__ __forceinline__ int imin(int a, int b) { return a < b ? a : b; }

__device__ __forceinline__ v8f wmma_bf16(const v16bf& a, const v16bf& b, const v8f& c) {
  return __builtin_amdgcn_wmma_f32_16x16x32_bf16(false, a, false, b, (short)0, c, false, false);
}

// ---------------- degree / norm ----------------
__global__ void k_deg_init(float* deg, int n) {
  int i = blockIdx.x * blockDim.x + threadIdx.x;
  if (i < n) deg[i] = 1.0f;                       // self-loop
}
__global__ void k_deg_acc(const int* __restrict__ dst, float* deg, int e) {
  int i = blockIdx.x * blockDim.x + threadIdx.x;
  if (i < e) atomicAdd(&deg[dst[i]], 1.0f);
}
__global__ void k_dinv(float* deg, int n) {
  int i = blockIdx.x * blockDim.x + threadIdx.x;
  if (i < n) deg[i] = rsqrtf(deg[i]);             // deg >= 1 always
}

// ---------------- conversions ----------------
__global__ void k_f32_to_bf16(const float* __restrict__ in, bf16_t* __restrict__ out, long total) {
  long i = blockIdx.x * (long)blockDim.x + threadIdx.x;
  if (i < total) out[i] = (bf16_t)in[i];
}
// W is K x Ncol row-major; write Wt as Ncol x K row-major (bf16)
__global__ void k_transpose_bf16(const float* __restrict__ W, bf16_t* __restrict__ Wt, int K, int Ncol) {
  long i = blockIdx.x * (long)blockDim.x + threadIdx.x;
  if (i < (long)K * Ncol) {
    int k = (int)(i / Ncol), c = (int)(i % Ncol);
    Wt[(long)c * K + k] = (bf16_t)W[i];
  }
}

// ---------------- WMMA NT GEMM with LDS-staged B ----------------
// C[M,Nc] = A[M,K] * Bt[Nc,K]^T ; M,Nc multiples of 16; K multiple of 128.
// Block = 8 waves computing a 128x64 macro-tile:
//   8 m-tiles (one per wave) x 4 n-tiles (4 named accumulators per wave).
// Shared 64xBKC bf16 B chunk staged in LDS via GLOBAL_LOAD_ASYNC_TO_LDS_B128.
// Out-of-range tiles are clamped (not early-returned) so all waves reach barriers
// and EXEC stays all-1s for WMMA. All 4 n-tiles are always computed (staging
// clamps columns, so LDS is fully valid); only stores are guarded.
__global__ __launch_bounds__(256) void k_gemm_nt_bf16_lds(
    const bf16_t* __restrict__ A, const bf16_t* __restrict__ Bt,
    float* __restrict__ C, int M, int Nc, int K,
    int apply_sigmoid, float out_scale) {
  __shared__ bf16_t sB[64 * BKC];                 // 16 KB

  const int lane   = threadIdx.x & 31;
  const int wave   = threadIdx.x >> 5;
  const int tilesM = M >> 4;
  const int tilesN = Nc >> 4;
  const int tm     = blockIdx.y * 8 + wave;
  const int tmc    = tm < tilesM ? tm : tilesM - 1;   // clamped for loads
  const int jt0    = blockIdx.x * 4;                  // first n-tile of this block
  const int njj    = imin(4, tilesN - jt0);           // block-uniform valid n-tiles

  const int rowA = (tmc << 4) + (lane & 15);
  const int kA   = (lane >> 4) << 4;              // 0 or 16
  const int kB   = (lane >> 4) << 4;              // 0 or 16
  const bf16_t* ap = A + (long)rowA * K;

  v8f acc0 = {}, acc1 = {}, acc2 = {}, acc3 = {};

  for (int kc = 0; kc < K; kc += BKC) {
    // ---- stage Bt[jt0*16 .. jt0*16+63][kc .. kc+BKC) into sB[64][BKC] ----
    // 64 rows x (BKC/8) 16-byte chunks = 1024 chunks, 4 per thread
    for (int t = threadIdx.x; t < 64 * (BKC / 8); t += 256) {
      int r  = t >> 4;                 // 0..63
      int ck = (t & 15) << 3;          // element offset within chunk row
      int col = imin(jt0 * 16 + r, Nc - 1);
      const bf16_t* gp = Bt + (long)col * K + kc + ck;
      bf16_t* lp = sB + r * BKC + ck;
#if __has_builtin(__builtin_amdgcn_global_load_async_to_lds_b128)
      __builtin_amdgcn_global_load_async_to_lds_b128(
          (gas_v4i*)(unsigned long long)gp,
          (las_v4i*)(unsigned long long)lp, 0, 0);
#else
      *(v8bf*)lp = *(const v8bf*)gp;
#endif
    }
#if __has_builtin(__builtin_amdgcn_s_wait_asynccnt)
    __builtin_amdgcn_s_wait_asynccnt(0);
#endif
    __syncthreads();

    // ---- accumulate: reuse A fragment across 4 independent n-tile accumulators ----
#pragma unroll
    for (int k0 = 0; k0 < BKC; k0 += 32) {
      __builtin_prefetch(ap + kc + k0 + 256, 0, 0);       // global_prefetch_b8
      v8bf a0 = *(const v8bf*)(ap + kc + k0 + kA);        // K = kA..kA+7
      v8bf a1 = *(const v8bf*)(ap + kc + k0 + 8 + kA);    // K = 8+kA..15+kA
      v16bf afr;
#pragma unroll
      for (int i = 0; i < 8; ++i) { afr[i] = a0[i]; afr[i + 8] = a1[i]; }

      const bf16_t* sbase = sB + (lane & 15) * BKC + k0 + kB;
      v16bf b0 = *(const v16bf*)(sbase + 0 * 16 * BKC);
      v16bf b1 = *(const v16bf*)(sbase + 1 * 16 * BKC);
      v16bf b2 = *(const v16bf*)(sbase + 2 * 16 * BKC);
      v16bf b3 = *(const v16bf*)(sbase + 3 * 16 * BKC);
      acc0 = wmma_bf16(afr, b0, acc0);
      acc1 = wmma_bf16(afr, b1, acc1);
      acc2 = wmma_bf16(afr, b2, acc2);
      acc3 = wmma_bf16(afr, b3, acc3);
    }
    __syncthreads();
  }

  // ---- epilogue (static unroll; stores guarded) ----
  if (tm < tilesM) {
    const int cm = (tm << 4) + ((lane >> 4) << 3);
    auto store_tile = [&](int J, const v8f& a) {
      if (J >= njj) return;
      const int cn = (jt0 + J) * 16 + (lane & 15);
#pragma unroll
      for (int r = 0; r < 8; ++r) {
        float v = a[r];
        if (apply_sigmoid) v = out_scale / (1.0f + __expf(-v));
        C[(long)(cm + r) * Nc + cn] = v;
      }
    };
    store_tile(0, acc0);
    store_tile(1, acc1);
    store_tile(2, acc2);
    store_tile(3, acc3);
  }
}

// ---------------- GCN aggregation ----------------
// Hacc[i,f] = Y[i,f]*dinv[i]^2 + bias[f]   (self-loop + bias; zero-inits accumulator)
__global__ void k_selfloop_bias(const float* __restrict__ Y, const float* __restrict__ dinv,
                                const float* __restrict__ b, float* __restrict__ Hacc,
                                int n, int f) {
  long idx = blockIdx.x * (long)blockDim.x + threadIdx.x;
  if (idx >= (long)n * f) return;
  int i = (int)(idx / f), feat = (int)(idx % f);
  float di = dinv[i];
  Hacc[idx] = Y[idx] * di * di + b[feat];
}
// Hacc[dst,f] += Y[src,f]*dinv[src]*dinv[dst]
__global__ void k_scatter(const float* __restrict__ Y, const int* __restrict__ src,
                          const int* __restrict__ dst, const float* __restrict__ dinv,
                          float* __restrict__ Hacc, int e, int f) {
  long idx = blockIdx.x * (long)blockDim.x + threadIdx.x;
  if (idx >= (long)e * f) return;
  int ed = (int)(idx / f), feat = (int)(idx % f);
  int s = src[ed], d = dst[ed];
  atomicAdd(&Hacc[(long)d * f + feat], Y[(long)s * f + feat] * dinv[s] * dinv[d]);
}

// ---------------- BatchNorm (train stats), coalesced partial sums ----------------
__global__ void k_bn_zero(float* sums, int f) {
  int i = blockIdx.x * blockDim.x + threadIdx.x;
  if (i < 2 * f) sums[i] = 0.f;
}
// Each block reduces BN_ROWS rows with coalesced reads; per-feature partials in LDS
// via ds_add_f32 (lane t -> feature t%f: consecutive banks, conflict-free).
__global__ __launch_bounds__(256) void k_bn_partial(const float* __restrict__ H,
                                                    float* __restrict__ sums,
                                                    int n, int f) {
  __shared__ float s[2 * HID];
  for (int i = threadIdx.x; i < 2 * f; i += 256) s[i] = 0.f;
  __syncthreads();
  const long base  = (long)blockIdx.x * BN_ROWS * f;
  const long total = (long)n * f;
  long lim = total - base;
  if (lim > (long)BN_ROWS * f) lim = (long)BN_ROWS * f;
  for (long t = threadIdx.x; t < lim; t += 256) {
    float v = H[base + t];
    int feat = (int)(t & (f - 1));               // f is a power of two; base % f == 0
    atomicAdd(&s[feat], v);
    atomicAdd(&s[f + feat], v * v);
  }
  __syncthreads();
  for (int i = threadIdx.x; i < 2 * f; i += 256) atomicAdd(&sums[i], s[i]);
}
__global__ void k_bn_finalize(const float* __restrict__ sums, const float* __restrict__ g,
                              const float* __restrict__ be, float* __restrict__ scale,
                              float* __restrict__ shift, int n, int f) {
  int i = blockIdx.x * blockDim.x + threadIdx.x;
  if (i >= f) return;
  float m   = sums[i] / (float)n;
  float var = sums[f + i] / (float)n - m * m;
  float sc  = g[i] * rsqrtf(var + BN_EPS);
  scale[i] = sc;
  shift[i] = be[i] - m * sc;
}
// out_bf16 = relu(H*scale + shift)
__global__ void k_bn_relu_bf16(const float* __restrict__ H, const float* __restrict__ scale,
                               const float* __restrict__ shift, bf16_t* __restrict__ out,
                               int n, int f) {
  long idx = blockIdx.x * (long)blockDim.x + threadIdx.x;
  if (idx >= (long)n * f) return;
  int feat = (int)(idx % f);
  float v = H[idx] * scale[feat] + shift[feat];
  out[idx] = (bf16_t)fmaxf(v, 0.f);
}

// ---------------- host ----------------
extern "C" void kernel_launch(void* const* d_in, const int* in_sizes, int n_in,
                              void* d_out, int out_size, void* d_ws, size_t ws_size,
                              hipStream_t stream) {
  const float* x   = (const float*)d_in[0];
  const int*   ei  = (const int*)d_in[1];
  const float* W0  = (const float*)d_in[2];
  const float* b0  = (const float*)d_in[3];
  const float* g0  = (const float*)d_in[4];
  const float* be0 = (const float*)d_in[5];
  const float* W1  = (const float*)d_in[6];
  const float* b1  = (const float*)d_in[7];
  const float* g1  = (const float*)d_in[8];
  const float* be1 = (const float*)d_in[9];
  const float* Wmu = (const float*)d_in[10];
  const float* bmu = (const float*)d_in[11];
  float* out = (float*)d_out;

  const int n = in_sizes[0] / F_IN;   // 10000
  const int e = in_sizes[1] / 2;      // 320000
  const int* src = ei;
  const int* dst = ei + e;

  // workspace carve-out (256B aligned)
  char* ws = (char*)d_ws;
  size_t off = 0;
  auto carve = [&](size_t bytes) -> void* {
    void* p = ws + off;
    off = (off + bytes + 255) & ~(size_t)255;
    return p;
  };
  float*  dinv  = (float*) carve((size_t)n * 4);
  bf16_t* Hbf   = (bf16_t*)carve((size_t)n * HID * 2);
  float*  Y     = (float*) carve((size_t)n * HID * 4);
  float*  Hacc  = (float*) carve((size_t)n * HID * 4);
  bf16_t* W0t   = (bf16_t*)carve((size_t)F_IN * HID * 2);
  bf16_t* W1t   = (bf16_t*)carve((size_t)HID * HID * 2);
  bf16_t* Wmt   = (bf16_t*)carve((size_t)HID * DOUT * 2);
  float*  sums  = (float*) carve(2 * HID * 4);
  float*  scale = (float*) carve(HID * 4);
  float*  shift = (float*) carve(HID * 4);
  bf16_t* Zbf   = (bf16_t*)carve((size_t)n * DOUT * 2);
  (void)ws_size; (void)n_in; (void)out_size;

  const int B = 256;
  auto g1d = [&](long total) { return dim3((unsigned)((total + B - 1) / B)); };

  // degrees -> dinv
  k_deg_init<<<g1d(n), B, 0, stream>>>(dinv, n);
  k_deg_acc <<<g1d(e), B, 0, stream>>>(dst, dinv, e);
  k_dinv    <<<g1d(n), B, 0, stream>>>(dinv, n);

  // bf16 conversions
  k_f32_to_bf16  <<<g1d((long)n * F_IN), B, 0, stream>>>(x, Hbf, (long)n * F_IN);
  k_transpose_bf16<<<g1d((long)F_IN * HID), B, 0, stream>>>(W0, W0t, F_IN, HID);
  k_transpose_bf16<<<g1d((long)HID * HID),  B, 0, stream>>>(W1, W1t, HID, HID);
  k_transpose_bf16<<<g1d((long)HID * DOUT), B, 0, stream>>>(Wmu, Wmt, HID, DOUT);

  const unsigned gy = (unsigned)((n / 16 + 7) / 8);
  const dim3 gemm_grid   ((HID / 16 + 3) / 4, gy);    // 8 x 79
  const dim3 gemm_grid_mu((DOUT / 16 + 3) / 4, gy);   // 2 x 79
  const unsigned bn_grid = (unsigned)((n + BN_ROWS - 1) / BN_ROWS);

  // ---- layer 0 ----
  k_gemm_nt_bf16_lds<<<gemm_grid, B, 0, stream>>>(Hbf, W0t, Y, n, HID, F_IN, 0, 1.0f);
  k_selfloop_bias<<<g1d((long)n * HID), B, 0, stream>>>(Y, dinv, b0, Hacc, n, HID);
  k_scatter<<<g1d((long)e * HID), B, 0, stream>>>(Y, src, dst, dinv, Hacc, e, HID);
  k_bn_zero<<<g1d(2 * HID), B, 0, stream>>>(sums, HID);
  k_bn_partial<<<bn_grid, B, 0, stream>>>(Hacc, sums, n, HID);
  k_bn_finalize<<<g1d(HID), B, 0, stream>>>(sums, g0, be0, scale, shift, n, HID);
  k_bn_relu_bf16<<<g1d((long)n * HID), B, 0, stream>>>(Hacc, scale, shift, Hbf, n, HID);

  // ---- layer 1 ----
  k_gemm_nt_bf16_lds<<<gemm_grid, B, 0, stream>>>(Hbf, W1t, Y, n, HID, HID, 0, 1.0f);
  k_selfloop_bias<<<g1d((long)n * HID), B, 0, stream>>>(Y, dinv, b1, Hacc, n, HID);
  k_scatter<<<g1d((long)e * HID), B, 0, stream>>>(Y, src, dst, dinv, Hacc, e, HID);
  k_bn_zero<<<g1d(2 * HID), B, 0, stream>>>(sums, HID);
  k_bn_partial<<<bn_grid, B, 0, stream>>>(Hacc, sums, n, HID);
  k_bn_finalize<<<g1d(HID), B, 0, stream>>>(sums, g1, be1, scale, shift, n, HID);
  k_bn_relu_bf16<<<g1d((long)n * HID), B, 0, stream>>>(Hacc, scale, shift, Hbf, n, HID);

  // ---- layer mu ----
  k_gemm_nt_bf16_lds<<<gemm_grid_mu, B, 0, stream>>>(Hbf, Wmt, Y, n, DOUT, HID, 0, 1.0f);
  k_selfloop_bias<<<g1d((long)n * DOUT), B, 0, stream>>>(Y, dinv, bmu, Hacc, n, DOUT);
  k_scatter<<<g1d((long)e * DOUT), B, 0, stream>>>(Y, src, dst, dinv, Hacc, e, DOUT);
  k_f32_to_bf16<<<g1d((long)n * DOUT), B, 0, stream>>>(Hacc, Zbf, (long)n * DOUT);

  // ---- decoder: out = sigmoid(Z Z^T) * adj_norm ----
  double nn = (double)n * (double)n;
  float adj_norm = (float)(nn / ((nn - (double)e) * 2.0));
  const dim3 zzt_grid((unsigned)((n / 16 + 3) / 4), gy); // 157 x 79
  k_gemm_nt_bf16_lds<<<zzt_grid, B, 0, stream>>>(Zbf, Zbf, out, n, n, DOUT, 1, adj_norm);
}
